// MultiheadAttention_63307817943830
// MI455X (gfx1250) — matrix operand order repaired and checked by reading, not compiled, not run on hardware
//
#include <hip/hip_runtime.h>

// MI455X / gfx1250 fused multi-head attention, bf16 WMMA pipeline.
// Workspace layout (bytes), total 75,497,472 (~72 MB):
//   [0,2M)    projT_q bf16 (H,64,D)
//   [2M,4M)   projT_k
//   [4M,6M)   projT_v
//   [6M,8M)   w_out  bf16 (1024,1024)
//   [8M,24M)  Qp  bf16 (B,H,N,64)
//   [24M,40M) Kp  bf16 (B,H,N,64)
//   [40M,56M) VpT bf16 (B,H,64,N)
//   [56M,72M) hid bf16 (B,N,H*64)

typedef __attribute__((ext_vector_type(16))) __bf16 v16bf;
typedef __attribute__((ext_vector_type(8)))  __bf16 v8bf;
typedef __attribute__((ext_vector_type(8)))  float  v8f;
typedef __attribute__((ext_vector_type(16))) float  v16f;

static constexpr int BB = 4;
static constexpr int NN = 2048;
static constexpr int DD = 1024;
static constexpr int HH = 16;
static constexpr int DKV = 64;

#define DEV static __device__ __forceinline__

DEV __bf16 f2bf(float x) { return (__bf16)x; }   // hardware cvt path

DEV v16bf pack16(v8bf lo, v8bf hi) {
  v16bf a;
#pragma unroll
  for (int i = 0; i < 8; ++i) { a[i] = lo[i]; a[i + 8] = hi[i]; }
  return a;
}

// A-fragment (16x32 bf16): lane (r,half) holds K = half*16+[0..15]
DEV v16bf load_a_bf16(const __bf16* row, int o1, int o2) {
  v8bf lo = *(const v8bf*)(row + o1);
  v8bf hi = *(const v8bf*)(row + o2);
  return pack16(lo, hi);
}

DEV v16bf load_a_f32(const float* row, int o1, int o2) {
  float4 f0 = *(const float4*)(row + o1);
  float4 f1 = *(const float4*)(row + o1 + 4);
  float4 f2 = *(const float4*)(row + o2);
  float4 f3 = *(const float4*)(row + o2 + 4);
  v16f t = {f0.x, f0.y, f0.z, f0.w, f1.x, f1.y, f1.z, f1.w,
            f2.x, f2.y, f2.z, f2.w, f3.x, f3.y, f3.z, f3.w};
  return __builtin_convertvector(t, v16bf);      // packed f32->bf16 conversion
}

#define WMMA_BF16(a, b, c) \
  __builtin_amdgcn_wmma_f32_16x16x32_bf16(false, (a), false, (b), (short)0, (c), false, false)

// ---------- weight prep ----------
// in: (H, D, 64) f32 -> out: (H, 64, D) bf16
__global__ __launch_bounds__(256) void k_transpose_bf16(const float* __restrict__ in,
                                                        unsigned short* __restrict__ outv) {
  __bf16* out = (__bf16*)outv;
  int idx = blockIdx.x * 256 + threadIdx.x;   // over H*64*D = 1048576
  int d  = idx & (DD - 1);
  int hn = idx >> 10;
  int n  = hn & 63;
  int h  = hn >> 6;
  out[idx] = f2bf(in[((size_t)h * DD + d) * 64 + n]);
}

__global__ __launch_bounds__(256) void k_convert_bf16(const float* __restrict__ in,
                                                      unsigned short* __restrict__ outv, int n) {
  __bf16* out = (__bf16*)outv;
  int idx = blockIdx.x * 256 + threadIdx.x;
  if (idx < n) out[idx] = f2bf(in[idx]);
}

// ---------- per-head projection GEMM (32x64 tile per wave) ----------
// X (B,N,D) f32 ; WT (H,64,D) bf16 ; bias (H,64) f32
// transposeOut=0: out (B,H,N,64) bf16   transposeOut=1: out (B,H,64,N) bf16
__global__ __launch_bounds__(256) void k_proj(const float* __restrict__ X,
                                              const unsigned short* __restrict__ WTv,
                                              const float* __restrict__ bias,
                                              unsigned short* __restrict__ outv,
                                              int transposeOut) {
  const __bf16* WT = (const __bf16*)WTv;
  __bf16* out = (__bf16*)outv;
  const int lane = threadIdx.x & 31, w = threadIdx.x >> 5;
  const int r = lane & 15, half = lane >> 4;
  int gw = blockIdx.x * 8 + w;         // 4096 waves total
  int mt = gw & 63;                    // N/32 row tiles per (b,h)
  int bh = gw >> 6;
  int b = bh >> 4, h = bh & 15;
  int m0 = mt * 32;

  v8f acc[2][4];
#pragma unroll
  for (int s = 0; s < 2; ++s)
#pragma unroll
    for (int t = 0; t < 4; ++t)
#pragma unroll
      for (int j = 0; j < 8; ++j) acc[s][t][j] = 0.f;

  const float* arow0 = X + ((size_t)b * NN + m0 + r) * DD;
  const float* arow1 = X + ((size_t)b * NN + m0 + 16 + r) * DD;
  const __bf16* wbase = WT + (size_t)h * 64 * DD;

  for (int kb = 0; kb < DD; kb += 32) {
    // issue all loads first, then burst 8 WMMAs (one wait, 8 matrix ops)
    v16bf a0 = load_a_f32(arow0, kb + half * 16, kb + half * 16 + 8);
    v16bf a1 = load_a_f32(arow1, kb + half * 16, kb + half * 16 + 8);
    v16bf bm[4];
#pragma unroll
    for (int t = 0; t < 4; ++t)
      bm[t] = *(const v16bf*)(wbase + (size_t)(t * 16 + r) * DD + kb + half * 16);
#pragma unroll
    for (int t = 0; t < 4; ++t) {
      acc[0][t] = WMMA_BF16(a0, bm[t], acc[0][t]);
      acc[1][t] = WMMA_BF16(a1, bm[t], acc[1][t]);
    }
  }
#pragma unroll
  for (int t = 0; t < 4; ++t) {
    float bv = bias[h * 64 + t * 16 + r];
#pragma unroll
    for (int s = 0; s < 2; ++s)
#pragma unroll
      for (int j = 0; j < 8; ++j) {
        int m = s * 16 + half * 8 + j;   // C layout: VGPR j -> row 8*half+j, lane r -> col
        float vvl = acc[s][t][j] + bv;
        if (!transposeOut)
          out[((size_t)bh * NN + m0 + m) * 64 + t * 16 + r] = f2bf(vvl);
        else
          out[((size_t)bh * 64 + t * 16 + r) * NN + m0 + m] = f2bf(vvl);
      }
  }
}

// ---------- fused flash attention ----------
// Qp,Kp (B,H,N,64) bf16 ; VpT (B,H,64,N) bf16 ; hid (B,N,H*64) bf16
__global__ __launch_bounds__(256) void k_flash(const unsigned short* __restrict__ Qpv,
                                               const unsigned short* __restrict__ Kpv,
                                               const unsigned short* __restrict__ VpTv,
                                               unsigned short* __restrict__ hidv) {
  const __bf16* Qp  = (const __bf16*)Qpv;
  const __bf16* Kp  = (const __bf16*)Kpv;
  const __bf16* VpT = (const __bf16*)VpTv;
  __bf16* hid = (__bf16*)hidv;

  __shared__ __bf16 pbuf[8][16 * 32];   // per-wave P-tile staging (C layout -> A layout)

  const int lane = threadIdx.x & 31, w = threadIdx.x >> 5;
  const int r = lane & 15, half = lane >> 4;
  int gw = blockIdx.x * 8 + w;          // 8192 waves: (b,h,qtile)
  int qt = gw & 127;
  int bh = gw >> 7;
  int b = bh >> 4, h = bh & 15;
  int q0 = qt * 16;

  const __bf16* qrow = Qp + ((size_t)bh * NN + q0 + r) * 64;
  v16bf qa0 = load_a_bf16(qrow, half * 16, half * 16 + 8);        // dk 0..31
  v16bf qa1 = load_a_bf16(qrow, 32 + half * 16, 40 + half * 16);   // dk 32..63

  v8f o[4];
#pragma unroll
  for (int t = 0; t < 4; ++t)
#pragma unroll
    for (int j = 0; j < 8; ++j) o[t][j] = 0.f;
  float mrow[8], lrow[8], al[8];
#pragma unroll
  for (int j = 0; j < 8; ++j) { mrow[j] = -1e30f; lrow[j] = 0.f; }

  const __bf16* kbase = Kp + (size_t)bh * NN * 64;
  const __bf16* vbase = VpT + (size_t)bh * 64 * NN;
  __bf16* pb = &pbuf[w][0];

  for (int j0 = 0; j0 < NN; j0 += 32) {
    // prefetch next key/value block into cache while we compute this one
    if (j0 + 32 < NN) {
      __builtin_prefetch(kbase + (size_t)(j0 + 32 + r) * 64, 0, 1);
      __builtin_prefetch(kbase + (size_t)(j0 + 48 + r) * 64, 0, 1);
      __builtin_prefetch(vbase + (size_t)(r * 4) * NN + j0 + 32, 0, 1);
    }
    // S = Q * K^T : load all 4 B fragments, then 4 WMMAs
    v8f s0, s1;
#pragma unroll
    for (int j = 0; j < 8; ++j) { s0[j] = 0.f; s1[j] = 0.f; }
    {
      const __bf16* k0 = kbase + (size_t)(j0 + r) * 64;
      const __bf16* k1 = kbase + (size_t)(j0 + 16 + r) * 64;
      v16bf bk00 = *(const v16bf*)(k0 + half * 16);
      v16bf bk01 = *(const v16bf*)(k0 + 32 + half * 16);
      v16bf bk10 = *(const v16bf*)(k1 + half * 16);
      v16bf bk11 = *(const v16bf*)(k1 + 32 + half * 16);
      s0 = WMMA_BF16(qa0, bk00, s0);
      s0 = WMMA_BF16(qa1, bk01, s0);
      s1 = WMMA_BF16(qa0, bk10, s1);
      s1 = WMMA_BF16(qa1, bk11, s1);
    }
    // online softmax: rows live in 16-lane groups; reduce with xor shuffles 1,2,4,8
#pragma unroll
    for (int j = 0; j < 8; ++j) {
      float a0 = s0[j] * 0.125f;   // /sqrt(64)
      float a1 = s1[j] * 0.125f;
      float t = fmaxf(a0, a1);
#pragma unroll
      for (int off = 1; off < 16; off <<= 1) t = fmaxf(t, __shfl_xor(t, off, 32));
      float mn = fmaxf(mrow[j], t);
      float alpha = __expf(mrow[j] - mn);
      float p0 = __expf(a0 - mn);
      float p1 = __expf(a1 - mn);
      float rs = p0 + p1;
#pragma unroll
      for (int off = 1; off < 16; off <<= 1) rs += __shfl_xor(rs, off, 32);
      lrow[j] = lrow[j] * alpha + rs;
      mrow[j] = mn;
      al[j] = alpha;
      pb[(half * 8 + j) * 32 + r]      = f2bf(p0);   // P[m][k] , k = key offset in block
      pb[(half * 8 + j) * 32 + 16 + r] = f2bf(p1);
    }
#pragma unroll
    for (int t4 = 0; t4 < 4; ++t4)
#pragma unroll
      for (int j = 0; j < 8; ++j) o[t4][j] *= al[j];

    // O += P(16x32) * V(32x64); VpT rows are key-contiguous
    v16bf pa = load_a_bf16(pb + r * 32, half * 16, half * 16 + 8);
    v16bf bv[4];
#pragma unroll
    for (int t4 = 0; t4 < 4; ++t4)
      bv[t4] = *(const v16bf*)(vbase + (size_t)(t4 * 16 + r) * NN + j0 + half * 16);
#pragma unroll
    for (int t4 = 0; t4 < 4; ++t4) o[t4] = WMMA_BF16(pa, bv[t4], o[t4]);
  }

  // normalize and write concat-head hidden state
#pragma unroll
  for (int t4 = 0; t4 < 4; ++t4)
#pragma unroll
    for (int j = 0; j < 8; ++j) {
      int m = half * 8 + j;
      float vvl = o[t4][j] / lrow[j];
      hid[((size_t)b * NN + q0 + m) * (HH * DKV) + h * 64 + t4 * 16 + r] = f2bf(vvl);
    }
}

// ---------- output projection: out(8192,1024) f32 = hid(8192,1024) @ w_out^T ----------
// 32x64 tile per wave
__global__ __launch_bounds__(256) void k_out(const unsigned short* __restrict__ hidv,
                                             const unsigned short* __restrict__ wv,
                                             float* __restrict__ out) {
  const __bf16* hid = (const __bf16*)hidv;
  const __bf16* wob = (const __bf16*)wv;
  const int lane = threadIdx.x & 31, w = threadIdx.x >> 5;
  const int r = lane & 15, half = lane >> 4;
  int gw = blockIdx.x * 8 + w;        // 4096 waves
  int rt = gw & 255;                  // 8192/32 row tiles
  int cg = gw >> 8;                   // 16 column groups of 64
  int r0 = rt * 32;

  v8f acc[2][4];
#pragma unroll
  for (int s = 0; s < 2; ++s)
#pragma unroll
    for (int t = 0; t < 4; ++t)
#pragma unroll
      for (int j = 0; j < 8; ++j) acc[s][t][j] = 0.f;

  const __bf16* arow0 = hid + (size_t)(r0 + r) * 1024;
  const __bf16* arow1 = hid + (size_t)(r0 + 16 + r) * 1024;
  for (int kb = 0; kb < 1024; kb += 32) {
    v16bf a0 = load_a_bf16(arow0, kb + half * 16, kb + half * 16 + 8);
    v16bf a1 = load_a_bf16(arow1, kb + half * 16, kb + half * 16 + 8);
    v16bf bm[4];
#pragma unroll
    for (int t = 0; t < 4; ++t)
      bm[t] = *(const v16bf*)(wob + (size_t)(cg * 64 + t * 16 + r) * 1024 + kb + half * 16);
#pragma unroll
    for (int t = 0; t < 4; ++t) {
      acc[0][t] = WMMA_BF16(a0, bm[t], acc[0][t]);
      acc[1][t] = WMMA_BF16(a1, bm[t], acc[1][t]);
    }
  }
#pragma unroll
  for (int s = 0; s < 2; ++s)
#pragma unroll
    for (int t = 0; t < 4; ++t)
#pragma unroll
      for (int j = 0; j < 8; ++j) {
        int m = s * 16 + half * 8 + j;
        out[(size_t)(r0 + m) * 1024 + cg * 64 + t * 16 + r] = acc[s][t][j];
      }
}

extern "C" void kernel_launch(void* const* d_in, const int* in_sizes, int n_in,
                              void* d_out, int out_size, void* d_ws, size_t ws_size,
                              hipStream_t stream) {
  const float* query  = (const float*)d_in[0];
  const float* key    = (const float*)d_in[1];
  const float* value  = (const float*)d_in[2];
  const float* proj_q = (const float*)d_in[3];
  const float* bias_q = (const float*)d_in[4];
  const float* proj_k = (const float*)d_in[5];
  const float* bias_k = (const float*)d_in[6];
  const float* proj_v = (const float*)d_in[7];
  const float* bias_v = (const float*)d_in[8];
  const float* w_out  = (const float*)d_in[9];

  char* ws = (char*)d_ws;
  const size_t SMALL = (size_t)1 << 21;       // 2 MB each weight buffer
  const size_t BIG   = (size_t)16777216;      // 16 MB each activation buffer
  unsigned short* pTq = (unsigned short*)(ws);
  unsigned short* pTk = (unsigned short*)(ws + SMALL);
  unsigned short* pTv = (unsigned short*)(ws + 2 * SMALL);
  unsigned short* wob = (unsigned short*)(ws + 3 * SMALL);
  unsigned short* Qp  = (unsigned short*)(ws + 4 * SMALL);
  unsigned short* Kp  = (unsigned short*)(ws + 4 * SMALL + BIG);
  unsigned short* VpT = (unsigned short*)(ws + 4 * SMALL + 2 * BIG);
  unsigned short* hid = (unsigned short*)(ws + 4 * SMALL + 3 * BIG);

  dim3 blk(256);
  k_transpose_bf16<<<4096, blk, 0, stream>>>(proj_q, pTq);
  k_transpose_bf16<<<4096, blk, 0, stream>>>(proj_k, pTk);
  k_transpose_bf16<<<4096, blk, 0, stream>>>(proj_v, pTv);
  k_convert_bf16<<<4096, blk, 0, stream>>>(w_out, wob, 1024 * 1024);

  k_proj<<<512, blk, 0, stream>>>(query, pTq, bias_q, Qp, 0);
  k_proj<<<512, blk, 0, stream>>>(key,   pTk, bias_k, Kp, 0);
  k_proj<<<512, blk, 0, stream>>>(value, pTv, bias_v, VpT, 1);

  k_flash<<<1024, blk, 0, stream>>>(Qp, Kp, VpT, hid);
  k_out<<<512, blk, 0, stream>>>(hid, wob, (float*)d_out);
}
